// Net_60198261621521
// MI455X (gfx1250) — matrix operand order repaired and checked by reading, compiled, not run on hardware
//
#include <hip/hip_runtime.h>
#include <math.h>

// ---------------------------------------------------------------------------
// GAT 2-layer forward for MI455X (gfx1250, wave32).
//   N=50000 nodes, E=1.6M edges (+N self loops), IN=512, H=8, C1=8, OUT=64.
// GEMMs use v_wmma_f32_16x16x32_bf16 with B pre-packed into the per-lane
// fragment layout (inner loop = 4x global_load_b128 + 1x v_wmma).
// Segment softmax/aggregation via L2-resident f32/u32 atomics.
// ---------------------------------------------------------------------------

typedef __bf16 bf16_t;
typedef bf16_t v16bf __attribute__((ext_vector_type(16)));
typedef float  v8f   __attribute__((ext_vector_type(8)));

union FragBF { v16bf v; unsigned short u[16]; };
struct alignas(16) US8 { unsigned short s[8]; };

__device__ __forceinline__ unsigned short f32_to_bf16_rne(float f) {
  unsigned int u = __float_as_uint(f);
  u += 0x7FFFu + ((u >> 16) & 1u);           // round-to-nearest-even
  return (unsigned short)(u >> 16);
}

// Order-preserving float<->uint encoding for atomicMax-based segment max.
__device__ __forceinline__ unsigned int enc_f32(float f) {
  unsigned int u = __float_as_uint(f);
  return (u & 0x80000000u) ? ~u : (u ^ 0x80000000u);
}
__device__ __forceinline__ float dec_f32(unsigned int k) {
  unsigned int u = (k & 0x80000000u) ? (k ^ 0x80000000u) : ~k;
  return __uint_as_float(u);
}

// --------------------------- utility kernels -------------------------------

__global__ void k_cvt_bf16(const float* __restrict__ src,
                           unsigned short* __restrict__ dst, int n) {
  int i = blockIdx.x * blockDim.x + threadIdx.x;
  if (i < n) dst[i] = f32_to_bf16_rne(src[i]);
}

__global__ void k_zero(unsigned int* __restrict__ p, int n) {
  int i = blockIdx.x * blockDim.x + threadIdx.x;
  if (i < n) p[i] = 0u;
}

// Pack B[K,Nc] (f32, row-major) into per-lane WMMA bf16 fragments:
// fragment f = (kb * (Nc/16) + ct); lane holds 16 contiguous bf16:
//   e[i] = B[(kb*32 + (lane>>4)*16 + i)*Nc + ct*16 + (lane&15)]
// One thread per (fragment, lane): 16 strided reads, one 32B packed write.
__global__ void k_pack_b(const float* __restrict__ B,
                         unsigned short* __restrict__ Bp, int K, int Nc) {
  int t = blockIdx.x * blockDim.x + threadIdx.x;
  int nct = Nc >> 4;
  int total = (K >> 5) * nct * 32;
  if (t >= total) return;
  int lane = t & 31, f = t >> 5;
  int kb = f / nct, ct = f % nct;
  int col = ct * 16 + (lane & 15);
  int krow = kb * 32 + (lane >> 4) * 16;
  unsigned short* o = Bp + (size_t)t * 16;
  #pragma unroll
  for (int i = 0; i < 16; ++i)
    o[i] = f32_to_bf16_rne(B[(size_t)(krow + i) * Nc + col]);
}

// --------------------------- WMMA GEMM -------------------------------------
// C[M,Nc] = A[M,K] @ B[K,Nc]; A bf16 row-major, Bp packed fragments (above),
// C f32 row-major. block = 128 threads = 4 wave32; wave w owns column tile w
// (Nc == 64); blockIdx.x owns a 16-row tile. Requires M%16==0, K%64==0, Nc==64.
__global__ void k_gemm_bf16_wmma(const unsigned short* __restrict__ A,
                                 const unsigned short* __restrict__ Bp,
                                 float* __restrict__ C,
                                 int M, int K, int Nc) {
  const int wave = threadIdx.x >> 5;
  const int lane = threadIdx.x & 31;
  const int lmod = lane & 15;
  const int lhi  = lane >> 4;                 // 0 / 1 half-wave
  const int rowBase = blockIdx.x * 16;
  const int colBase = wave * 16;
  const int nct = Nc >> 4;

  const unsigned short* aRow = A + (size_t)(rowBase + lmod) * K + lhi * 16;
  // packed B: fragment (kb*nct + wave), this lane's 16 values are contiguous
  const unsigned short* bFrag = Bp + (size_t)((0 * nct + wave) * 32 + lane) * 16;
  const size_t bStep = (size_t)nct * 32 * 16;  // advance one K-block (kb+1)

  v8f acc = {};
  #pragma unroll 2
  for (int k0 = 0; k0 < K; k0 += 32) {
    FragBF fa;                                 // A 16x32 bf16 fragment
    *(US8*)&fa.u[0] = *(const US8*)(aRow + k0);
    *(US8*)&fa.u[8] = *(const US8*)(aRow + k0 + 8);
    FragBF fb;                                 // B fragment: 2x b128 contiguous
    const unsigned short* bp = bFrag + (size_t)(k0 >> 5) * bStep;
    *(US8*)&fb.u[0] = *(const US8*)(bp);
    *(US8*)&fb.u[8] = *(const US8*)(bp + 8);
    acc = __builtin_amdgcn_wmma_f32_16x16x32_bf16(
        false, fa.v, false, fb.v, (short)0, acc, false, false);
  }

  // C/D layout: VGPR i -> row rowBase+i (lanes 0-15) / rowBase+8+i (lanes 16-31)
  const int rout = rowBase + lhi * 8;
  #pragma unroll
  for (int i = 0; i < 8; ++i)
    C[(size_t)(rout + i) * Nc + colBase + lmod] = acc[i];
}

// --------------------------- attention scalars -----------------------------
// as[n,h] = <h[n,h,:], a_src[h,:]>, ad likewise. One thread per (node, head).
__global__ void k_alpha(const float* __restrict__ h,
                        const float* __restrict__ a_src,
                        const float* __restrict__ a_dst,
                        float* __restrict__ as, float* __restrict__ ad,
                        int n, int heads, int ch) {
  int t = blockIdx.x * blockDim.x + threadIdx.x;
  if (t >= n * heads) return;
  int node = t / heads, hd = t % heads;
  const float* hp = h + (size_t)node * heads * ch + (size_t)hd * ch;
  const float* sp = a_src + hd * ch;
  const float* dp = a_dst + hd * ch;
  float vs = 0.f, vd = 0.f;
  for (int c = 0; c < ch; ++c) { vs += hp[c] * sp[c]; vd += hp[c] * dp[c]; }
  as[t] = vs; ad[t] = vd;
}

// --------------------------- edge passes -----------------------------------
// Pass 1: segment max of leaky_relu(as[src]+ad[dst]) per (dst, head).
__global__ void k_edge_max(const long long* __restrict__ ei, int E, int n,
                           const float* __restrict__ as,
                           const float* __restrict__ ad,
                           unsigned int* __restrict__ m, int heads) {
  int e = blockIdx.x * blockDim.x + threadIdx.x;
  if (e >= E + n) return;
  int s, d;
  if (e < E) { s = (int)ei[e]; d = (int)ei[(size_t)E + e]; }
  else       { s = d = e - E; }                    // self loop
  for (int hd = 0; hd < heads; ++hd) {
    float v = as[s * heads + hd] + ad[d * heads + hd];
    v = v > 0.f ? v : 0.2f * v;                    // leaky_relu slope 0.2
    atomicMax(&m[d * heads + hd], enc_f32(v));
  }
}

// Pass 2: ex = exp(e - max); denom[dst,h] += ex; acc[dst,h,:] += ex*h[src,h,:].
__global__ void k_edge_acc(const long long* __restrict__ ei, int E, int n,
                           const float* __restrict__ as,
                           const float* __restrict__ ad,
                           const unsigned int* __restrict__ m,
                           float* __restrict__ denom,
                           float* __restrict__ accum,
                           const float* __restrict__ h, int heads, int ch) {
  int e = blockIdx.x * blockDim.x + threadIdx.x;
  if (e >= E + n) return;
  int s, d;
  if (e < E) { s = (int)ei[e]; d = (int)ei[(size_t)E + e]; }
  else       { s = d = e - E; }
  for (int hd = 0; hd < heads; ++hd) {
    float v = as[s * heads + hd] + ad[d * heads + hd];
    v = v > 0.f ? v : 0.2f * v;
    float ex = __expf(v - dec_f32(m[d * heads + hd]));
    atomicAdd(&denom[d * heads + hd], ex);
    const float* hp = h     + (size_t)s * heads * ch + (size_t)hd * ch;
    float*       ap = accum + (size_t)d * heads * ch + (size_t)hd * ch;
    for (int c = 0; c < ch; ++c) atomicAdd(&ap[c], ex * hp[c]);
  }
}

// --------------------------- layer finalization ----------------------------
// Layer 1: y = bf16(elu(acc/denom + b1)) feeding GEMM2 as A.
__global__ void k_finalize1(const float* __restrict__ acc,
                            const float* __restrict__ denom,
                            const float* __restrict__ b,
                            unsigned short* __restrict__ yb,
                            int n, int heads, int ch) {
  int t = blockIdx.x * blockDim.x + threadIdx.x;
  if (t >= n * heads * ch) return;
  int hc = heads * ch;
  int node = t / hc, j = t % hc, hd = j / ch;
  float v = acc[t] / (denom[node * heads + hd] + 1e-16f) + b[j];
  v = v > 0.f ? v : (__expf(v) - 1.f);             // ELU
  yb[t] = f32_to_bf16_rne(v);
}

// Layer 2 (heads==1): z = acc/denom + b2 then log_softmax over 64 columns.
// One wave32 per node, 2 columns per lane, butterfly shuffles for max/sum.
__global__ void k_finalize2_logsoftmax(const float* __restrict__ acc,
                                       const float* __restrict__ denom,
                                       const float* __restrict__ b,
                                       float* __restrict__ out, int n, int nc) {
  int node = blockIdx.x * (blockDim.x >> 5) + (threadIdx.x >> 5);
  int lane = threadIdx.x & 31;
  if (node >= n) return;
  float dn = denom[node] + 1e-16f;
  float z0 = acc[(size_t)node * nc + lane]      / dn + b[lane];
  float z1 = acc[(size_t)node * nc + lane + 32] / dn + b[lane + 32];
  float mx = fmaxf(z0, z1);
  #pragma unroll
  for (int off = 16; off; off >>= 1) mx = fmaxf(mx, __shfl_xor(mx, off, 32));
  float se = __expf(z0 - mx) + __expf(z1 - mx);
  #pragma unroll
  for (int off = 16; off; off >>= 1) se += __shfl_xor(se, off, 32);
  float l = __logf(se) + mx;
  out[(size_t)node * nc + lane]      = z0 - l;
  out[(size_t)node * nc + lane + 32] = z1 - l;
}

// ---------------------------------------------------------------------------

extern "C" void kernel_launch(void* const* d_in, const int* in_sizes, int n_in,
                              void* d_out, int out_size, void* d_ws, size_t ws_size,
                              hipStream_t stream) {
  (void)in_sizes; (void)n_in; (void)out_size; (void)ws_size;
  constexpr int N   = 50000;
  constexpr int IN  = 512;
  constexpr int H1  = 8, C1 = 8, HC = H1 * C1;     // 64
  constexpr int OUT = 64;
  constexpr int E   = 1600000;
  constexpr int ET  = E + N;                       // with self loops

  const float*     x    = (const float*)d_in[0];
  const long long* ei   = (const long long*)d_in[1];   // int64 [2,E]
  const float*     W1   = (const float*)d_in[2];
  const float*     aS1  = (const float*)d_in[3];
  const float*     aD1  = (const float*)d_in[4];
  const float*     b1   = (const float*)d_in[5];
  const float*     W2   = (const float*)d_in[6];
  const float*     aS2  = (const float*)d_in[7];
  const float*     aD2  = (const float*)d_in[8];
  const float*     b2   = (const float*)d_in[9];
  float*           out  = (float*)d_out;

  // Workspace layout (bytes). xb region (51.2MB) is dead after GEMM1 and is
  // recycled for all layer-2 buffers (needs only 26.4MB of it).
  char* ws = (char*)d_ws;
  unsigned short* xb   = (unsigned short*)(ws + 0);           // N*IN bf16
  float*          h2   = (float*)(ws + 0);                    // N*64 f32 (reuse)
  float*          acc2 = (float*)(ws + 12800000);             // N*64 f32
  unsigned int*   m2   = (unsigned int*)(ws + 25600000);      // N u32
  float*          d2   = (float*)(ws + 25800000);             // N f32
  float*          as2  = (float*)(ws + 26000000);             // N f32
  float*          ad2  = (float*)(ws + 26200000);             // N f32
  unsigned short* W1p  = (unsigned short*)(ws + 51200000);    // 512*64 bf16 packed
  unsigned short* W2p  = (unsigned short*)(ws + 51265536);    // 64*64 bf16 packed
  float*          h1   = (float*)(ws + 51273728);             // N*64 f32
  float*          as1  = (float*)(ws + 64073728);             // N*8 f32
  float*          ad1  = (float*)(ws + 65673728);             // N*8 f32
  unsigned int*   m1   = (unsigned int*)(ws + 67273728);      // N*8 u32
  float*          d1   = (float*)(ws + 68873728);             // N*8 f32
  float*          acc1 = (float*)(ws + 70473728);             // N*64 f32
  unsigned short* y1b  = (unsigned short*)(ws + 83273728);    // N*64 bf16

  auto blks = [](int n, int t) { return (n + t - 1) / t; };

  // --- precision conversion / weight fragment packing ---
  k_cvt_bf16<<<blks(N * IN, 256), 256, 0, stream>>>(x, xb, N * IN);
  k_pack_b<<<blks((IN / 32) * (HC / 16) * 32, 256), 256, 0, stream>>>(W1, W1p, IN, HC);
  k_pack_b<<<blks((HC / 32) * (OUT / 16) * 32, 256), 256, 0, stream>>>(W2, W2p, HC, OUT);

  // ================= layer 1 =================
  k_gemm_bf16_wmma<<<N / 16, 128, 0, stream>>>(xb, W1p, h1, N, IN, HC);
  k_alpha<<<blks(N * H1, 256), 256, 0, stream>>>(h1, aS1, aD1, as1, ad1, N, H1, C1);
  // zero m1|d1|acc1 (contiguous 16MB)
  k_zero<<<blks(4000000, 256), 256, 0, stream>>>(m1, 4000000);
  k_edge_max<<<blks(ET, 256), 256, 0, stream>>>(ei, E, N, as1, ad1, m1, H1);
  k_edge_acc<<<blks(ET, 256), 256, 0, stream>>>(ei, E, N, as1, ad1, m1, d1, acc1, h1, H1, C1);
  k_finalize1<<<blks(N * HC, 256), 256, 0, stream>>>(acc1, d1, b1, y1b, N, H1, C1);

  // ================= layer 2 =================
  k_gemm_bf16_wmma<<<N / 16, 128, 0, stream>>>(y1b, W2p, h2, N, HC, OUT);
  // zero acc2|m2|d2 (contiguous 13.2MB)
  k_zero<<<blks(3300000, 256), 256, 0, stream>>>((unsigned int*)acc2, 3300000);
  k_alpha<<<blks(N, 256), 256, 0, stream>>>(h2, aS2, aD2, as2, ad2, N, 1, OUT);
  k_edge_max<<<blks(ET, 256), 256, 0, stream>>>(ei, E, N, as2, ad2, m2, 1);
  k_edge_acc<<<blks(ET, 256), 256, 0, stream>>>(ei, E, N, as2, ad2, m2, d2, acc2, h2, 1, OUT);
  k_finalize2_logsoftmax<<<blks(N * 32, 256), 256, 0, stream>>>(acc2, d2, b2, out, N, OUT);
}
